// BasicCNN_Detect_37718402793616
// MI455X (gfx1250) — matrix operand order repaired and checked by reading, not compiled, not run on hardware
//
#include <hip/hip_runtime.h>
#include <stdint.h>
#include <stddef.h>

typedef __attribute__((ext_vector_type(16))) _Float16 v16h;
typedef __attribute__((ext_vector_type(8)))  _Float16 v8h;
typedef __attribute__((ext_vector_type(8)))  float    v8f;

#define BATCH  1024
#define CHUNK  256
#define NCHUNK 4

// ---------------------------------------------------------------------------
// deterministic hash RNG -> standard normal (Box-Muller)
// ---------------------------------------------------------------------------
__device__ inline uint32_t mix3(uint32_t a, uint32_t b, uint32_t c) {
    uint32_t h = a * 0x9E3779B1u + b * 0x85EBCA77u + c * 0xC2B2AE3Du + 0x27D4EB2Fu;
    h ^= h >> 15; h *= 0x2545F491u;
    h ^= h >> 13; h *= 0x165667B1u;
    h ^= h >> 16;
    return h;
}

__device__ inline float nrm(uint32_t seed, uint32_t tensor, uint32_t e) {
    uint32_t h1 = mix3(seed, tensor, e * 2u);
    uint32_t h2 = mix3(seed, tensor, e * 2u + 1u);
    float u1 = (float)(h1 >> 8) * (1.0f / 16777216.0f) + (1.0f / 33554432.0f); // (0,1]
    float u2 = (float)(h2 >> 8) * (1.0f / 16777216.0f);
    return sqrtf(-2.0f * __logf(u1)) * __cosf(6.28318530717958647692f * u2);
}

// ---------------------------------------------------------------------------
// Bayesian weight sampling for one stochastic pass.
// B-operands are written TRANSPOSED ([N][K], K-contiguous) so the WMMA B
// fragment (one column's 16 K-values per lane) is two b128 loads per lane.
//   w2t  : [64][512]  conv2 B^T  (n = out ch < 50, k = i*25+q < 500)
//   f1wt : [512][800] fc1  B^T  (n = out unit < 500, k = input < 800)
// ---------------------------------------------------------------------------
#define SAMPLE_TOT (512*800 + 64*512 + 5000 + 500 + 500 + 50 + 20 + 10)

__global__ void sample_weights(uint32_t seed,
    const float* __restrict__ w1mu, const float* __restrict__ w1ls,
    const float* __restrict__ b1mu, const float* __restrict__ b1ls,
    const float* __restrict__ w2mu, const float* __restrict__ w2ls,
    const float* __restrict__ b2mu, const float* __restrict__ b2ls,
    const float* __restrict__ f1wmu, const float* __restrict__ f1wls,
    const float* __restrict__ f1bmu, const float* __restrict__ f1bls,
    const float* __restrict__ f2wmu, const float* __restrict__ f2wls,
    const float* __restrict__ f2bmu, const float* __restrict__ f2bls,
    float* __restrict__ w1f, float* __restrict__ b1f,
    _Float16* __restrict__ w2t, float* __restrict__ b2f,
    _Float16* __restrict__ f1wt, float* __restrict__ f1bf,
    float* __restrict__ f2wf, float* __restrict__ f2bf)
{
    int idx = blockIdx.x * blockDim.x + threadIdx.x;
    if (idx >= SAMPLE_TOT) return;

    if (idx < 512*800) {                    // fc1 W^T: row c (out unit), col r (input)
        int c = idx / 800, r = idx % 800;
        _Float16 v = (_Float16)0.0f;
        if (c < 500) { int e = r*500 + c; v = (_Float16)(f1wmu[e] + __expf(f1wls[e]) * nrm(seed, 4u, (uint32_t)e)); }
        f1wt[idx] = v; return;
    }
    idx -= 512*800;
    if (idx < 64*512) {                     // conv2 W^T: row n (out ch), col k
        int n = idx >> 9, k = idx & 511;
        _Float16 v = (_Float16)0.0f;
        if (k < 500 && n < 50) { int e = n*500 + k; v = (_Float16)(w2mu[e] + __expf(w2ls[e]) * nrm(seed, 2u, (uint32_t)e)); }
        w2t[idx] = v; return;
    }
    idx -= 64*512;
    if (idx < 5000) { f2wf[idx] = f2wmu[idx] + __expf(f2wls[idx]) * nrm(seed, 6u, (uint32_t)idx); return; }
    idx -= 5000;
    if (idx < 500)  { w1f[idx]  = w1mu[idx]  + __expf(w1ls[idx])  * nrm(seed, 0u, (uint32_t)idx); return; }
    idx -= 500;
    if (idx < 500)  { f1bf[idx] = f1bmu[idx] + __expf(f1bls[idx]) * nrm(seed, 5u, (uint32_t)idx); return; }
    idx -= 500;
    if (idx < 50)   { b2f[idx]  = b2mu[idx]  + __expf(b2ls[idx])  * nrm(seed, 3u, (uint32_t)idx); return; }
    idx -= 50;
    if (idx < 20)   { b1f[idx]  = b1mu[idx]  + __expf(b1ls[idx])  * nrm(seed, 1u, (uint32_t)idx); return; }
    idx -= 20;
    if (idx < 10)   { f2bf[idx] = f2bmu[idx] + __expf(f2bls[idx]) * nrm(seed, 7u, (uint32_t)idx); return; }
}

// ---------------------------------------------------------------------------
// WMMA GEMM:  C[M,N] (+)= A[M,K](f16 row-major) * Bt[N,K]^T (f16, N-major)
// One wave computes a 16x64 tile: one A fragment feeds 4 v_wmma per k-step.
// Per k-step VMEM: A = 2x b128, B = 8x b128  ->  10 loads : 4 WMMA.
// A/B use contiguous k order: lane half h holds k 16h..16h+15 (wave32). All loads 32B-aligned.
// ---------------------------------------------------------------------------
__global__ __launch_bounds__(128)
void wmma_gemm_f16f32(const _Float16* __restrict__ A, const _Float16* __restrict__ Bt,
                      float* __restrict__ C, int M, int N, int K,
                      int lda, int ldb, int ldc, int accumulate)
{
    const int lane = threadIdx.x & 31;
    const int wave = blockIdx.x * (blockDim.x >> 5) + (threadIdx.x >> 5);
    const int nG = N >> 6;                   // 64-column groups (N % 64 == 0)
    const int mt = wave / nG, ng = wave % nG;
    if (mt * 16 >= M) return;                // uniform per wave: EXEC stays all-ones
    const int row0 = mt << 4, colG = ng << 6;
    const int l15  = lane & 15;              // A row / B col / C col within tile
    const int ahk  = (lane >> 4) << 4;       // A: lanes 16-31 hold k+16..k+31
    const int bko  = (lane >> 4) << 4;       // B: lanes 16-31 hold k+16..k+31
    const int crow = (lane >> 4) << 3;       // C: lanes 16-31 hold rows 8..15

    v8f a0, a1, a2, a3;
    {
        const float* cp = C + (size_t)(row0 + crow) * ldc + colG + l15;
        #pragma unroll
        for (int r = 0; r < 8; ++r) {
            a0[r] = accumulate ? cp[(size_t)r * ldc +  0] : 0.0f;
            a1[r] = accumulate ? cp[(size_t)r * ldc + 16] : 0.0f;
            a2[r] = accumulate ? cp[(size_t)r * ldc + 32] : 0.0f;
            a3[r] = accumulate ? cp[(size_t)r * ldc + 48] : 0.0f;
        }
    }

    #pragma unroll 2
    for (int k0 = 0; k0 < K; k0 += 32) {
        const _Float16* ap = A + (size_t)(row0 + l15) * lda + k0 + ahk;
        v8h alo = *(const v8h*)(ap);
        v8h ahi = *(const v8h*)(ap + 8);
        v16h a = __builtin_shufflevector(alo, ahi, 0,1,2,3,4,5,6,7,8,9,10,11,12,13,14,15);
        __builtin_prefetch(ap + 64, 0, 0);   // stream A two k-blocks ahead

        const _Float16* bp = Bt + (size_t)(colG + l15) * ldb + k0 + bko;
        v16h b0 = *(const v16h*)(bp);
        v16h b1 = *(const v16h*)(bp + (size_t)16 * ldb);
        v16h b2 = *(const v16h*)(bp + (size_t)32 * ldb);
        v16h b3 = *(const v16h*)(bp + (size_t)48 * ldb);

        a0 = __builtin_amdgcn_wmma_f32_16x16x32_f16(false, a, false, b0, (short)0, a0, false, false);
        a1 = __builtin_amdgcn_wmma_f32_16x16x32_f16(false, a, false, b1, (short)0, a1, false, false);
        a2 = __builtin_amdgcn_wmma_f32_16x16x32_f16(false, a, false, b2, (short)0, a2, false, false);
        a3 = __builtin_amdgcn_wmma_f32_16x16x32_f16(false, a, false, b3, (short)0, a3, false, false);
    }

    float* cp = C + (size_t)(row0 + crow) * ldc + colG + l15;
    #pragma unroll
    for (int r = 0; r < 8; ++r) {
        cp[(size_t)r * ldc +  0] = a0[r];
        cp[(size_t)r * ldc + 16] = a1[r];
        cp[(size_t)r * ldc + 32] = a2[r];
        cp[(size_t)r * ldc + 48] = a3[r];
    }
}

// ---------------------------------------------------------------------------
// conv1 (1ch, 5x5, VALID) + ReLU + 2x2 maxpool fused, f16 output [B,20,12,12]
// ---------------------------------------------------------------------------
__global__ void conv1_relu_pool(const float* __restrict__ x, const float* __restrict__ w1,
                                const float* __restrict__ b1, _Float16* __restrict__ h1)
{
    int idx = blockIdx.x * blockDim.x + threadIdx.x;
    if (idx >= BATCH * 20 * 144) return;
    int ox = idx % 12, t = idx / 12;
    int oy = t % 12;   t /= 12;
    int c  = t % 20;   int b = t / 20;

    float w[25];
    #pragma unroll
    for (int i = 0; i < 25; ++i) w[i] = w1[c * 25 + i];
    const float* xb = x + (size_t)b * 784;
    float bias = b1[c];
    float mx = -3.4e38f;
    #pragma unroll
    for (int dy = 0; dy < 2; ++dy)
    #pragma unroll
    for (int dx = 0; dx < 2; ++dx) {
        int cy = 2 * oy + dy, cx = 2 * ox + dx;
        float acc = bias;
        #pragma unroll
        for (int ky = 0; ky < 5; ++ky)
            #pragma unroll
            for (int kx = 0; kx < 5; ++kx)
                acc = fmaf(xb[(cy + ky) * 28 + cx + kx], w[ky * 5 + kx], acc);
        mx = fmaxf(mx, acc);
    }
    h1[idx] = (_Float16)fmaxf(mx, 0.0f);     // relu(max) == max(relu)
}

// ---------------------------------------------------------------------------
// im2col for conv2 over one batch-chunk: rows = CHUNK*64, cols = 512 (500 valid)
// ---------------------------------------------------------------------------
__global__ void im2col_conv2(const _Float16* __restrict__ h1, _Float16* __restrict__ imc, int chunk)
{
    int idx = blockIdx.x * blockDim.x + threadIdx.x;
    if (idx >= CHUNK * 64 * 20) return;
    int c = idx % 20, r = idx / 20;
    int p = r & 63;
    int b = chunk * CHUNK + (r >> 6);
    int oy = p >> 3, ox = p & 7;
    const _Float16* src = h1 + ((size_t)(b * 20 + c) * 12 + oy) * 12 + ox;
    _Float16* dst = imc + (size_t)r * 512 + c * 25;
    #pragma unroll
    for (int ky = 0; ky < 5; ++ky)
        #pragma unroll
        for (int kx = 0; kx < 5; ++kx)
            dst[ky * 5 + kx] = src[ky * 12 + kx];
    if (c == 0) {
        _Float16* z = imc + (size_t)r * 512 + 500;
        #pragma unroll
        for (int q = 0; q < 12; ++q) z[q] = (_Float16)0.0f;
    }
}

// conv2 epilogue A: +bias, ReLU, 2x2 maxpool -> pooled [B,800] f16 (fc1 A-matrix)
__global__ void conv2_pool_epilogue(const float* __restrict__ c2raw, const float* __restrict__ b2,
                                    _Float16* __restrict__ pooled, int chunk)
{
    int idx = blockIdx.x * blockDim.x + threadIdx.x;
    if (idx >= CHUNK * 50 * 16) return;
    int pp = idx % 16, t = idx / 16;
    int c = t % 50, bloc = t / 50;
    int py = pp >> 2, px = pp & 3;
    float mx = -3.4e38f;
    #pragma unroll
    for (int dy = 0; dy < 2; ++dy)
    #pragma unroll
    for (int dx = 0; dx < 2; ++dx) {
        int p = (2 * py + dy) * 8 + (2 * px + dx);
        mx = fmaxf(mx, c2raw[(size_t)(bloc * 64 + p) * 64 + c]);
    }
    int b = chunk * CHUNK + bloc;
    pooled[(size_t)b * 800 + c * 16 + pp] = (_Float16)fmaxf(mx + b2[c], 0.0f);
}

// conv2 epilogue B: +bias only (pre-ReLU features) -> feat [B,3200] f16 (PCA A-matrix)
__global__ void conv2_feat_epilogue(const float* __restrict__ c2raw, const float* __restrict__ b2,
                                    _Float16* __restrict__ feat, int chunk)
{
    int idx = blockIdx.x * blockDim.x + threadIdx.x;
    if (idx >= CHUNK * 64 * 50) return;
    int c = idx % 50, r = idx / 50;
    int p = r & 63;
    int b = chunk * CHUNK + (r >> 6);
    feat[(size_t)b * 3200 + c * 64 + p] = (_Float16)(c2raw[(size_t)r * 64 + c] + b2[c]);
}

// fc1 epilogue: +bias, ReLU, f16, zero the 12 padded columns
__global__ void fc1_epilogue(const float* __restrict__ raw, const float* __restrict__ bias,
                             _Float16* __restrict__ f)
{
    int idx = blockIdx.x * blockDim.x + threadIdx.x;
    if (idx >= BATCH * 512) return;
    int j = idx & 511;
    float v = 0.0f;
    if (j < 500) v = fmaxf(raw[idx] + bias[j], 0.0f);
    f[idx] = (_Float16)v;
}

// fc2 (N=10, too narrow for WMMA): logits + optional aggregation
__global__ void fc2_bayes(const _Float16* __restrict__ f, const float* __restrict__ w,
                          const float* __restrict__ bias, float* __restrict__ out, float* agg)
{
    int idx = blockIdx.x * blockDim.x + threadIdx.x;
    if (idx >= BATCH * 10) return;
    int b = idx / 10, j = idx % 10;
    const _Float16* fb = f + (size_t)b * 512;
    float acc = bias[j];
    for (int k = 0; k < 500; ++k) acc = fmaf((float)fb[k], w[k * 10 + j], acc);
    out[idx] = acc;
    if (agg) agg[idx] += acc;
}

// ---------------------------------------------------------------------------
// misc small kernels
// ---------------------------------------------------------------------------
__global__ void zero_f32(float* p, int n)
{ int i = blockIdx.x * blockDim.x + threadIdx.x; if (i < n) p[i] = 0.0f; }

// pca_proj [12800,256] f32 -> projt [256,12800] f16 (B^T for the PCA GEMM)
__global__ void transpose_proj_f16(const float* __restrict__ in, _Float16* __restrict__ out)
{
    int idx = blockIdx.x * blockDim.x + threadIdx.x;
    if (idx >= 12800 * 256) return;
    int j = idx / 12800, i = idx % 12800;
    out[idx] = (_Float16)in[(size_t)i * 256 + j];
}

__global__ void pca_mean_proj(const float* __restrict__ mean, const float* __restrict__ proj,
                              float* __restrict__ pm)
{
    int j = threadIdx.x;                        // 256 threads, 1 block
    float acc = 0.0f;
    for (int k = 0; k < 12800; ++k) acc = fmaf(mean[k], proj[(size_t)k * 256 + j], acc);
    pm[j] = acc;
}

__global__ void fa3_init(float* __restrict__ fa3, const float* __restrict__ pm)
{ int i = blockIdx.x * blockDim.x + threadIdx.x; if (i < BATCH * 256) fa3[i] = -pm[i & 255]; }

__global__ void argmax10(const float* __restrict__ agg, int* __restrict__ pred)
{
    int b = blockIdx.x * blockDim.x + threadIdx.x;
    if (b >= BATCH) return;
    float best = agg[b * 10]; int bi = 0;
    for (int j = 1; j < 10; ++j) { float v = agg[b * 10 + j]; if (v > best) { best = v; bi = j; } }
    pred[b] = bi;
}

// ---------------------------------------------------------------------------
// LDS bitonic sorts (ascending) — n == blockDim.x, power of 2
// ---------------------------------------------------------------------------
__device__ inline void bitonic_sort(volatile float* s, int n, int tid)
{
    for (int k = 2; k <= n; k <<= 1) {
        for (int j = k >> 1; j > 0; j >>= 1) {
            int ixj = tid ^ j;
            if (ixj > tid) {
                float a = s[tid], b = s[ixj];
                bool up = ((tid & k) == 0);
                if ((a > b) == up) { s[tid] = b; s[ixj] = a; }
            }
            __syncthreads();
        }
    }
}

__global__ void sort_rows_256(const float* __restrict__ in, float* __restrict__ out)
{
    __shared__ float s[256];
    int row = blockIdx.x, tid = threadIdx.x;
    s[tid] = in[(size_t)row * 256 + tid];
    __syncthreads();
    bitonic_sort(s, 256, tid);
    out[(size_t)row * 256 + tid] = s[tid];
}

__global__ void sort_rows_40(const float* __restrict__ in, float* __restrict__ out)
{
    __shared__ float s[64];
    int row = blockIdx.x, tid = threadIdx.x;
    s[tid] = (tid < 40) ? in[(size_t)row * 40 + tid] : 3.4e38f;
    __syncthreads();
    bitonic_sort(s, 64, tid);
    if (tid < 40) out[(size_t)row * 40 + tid] = s[tid];
}

// res6 = tile(last_logits, 4) then sort row
__global__ void build_sort_res6(const float* __restrict__ last, float* __restrict__ out)
{
    __shared__ float s[64];
    int b = blockIdx.x, tid = threadIdx.x;
    s[tid] = (tid < 40) ? last[b * 10 + (tid % 10)] : 3.4e38f;
    __syncthreads();
    bitonic_sort(s, 64, tid);
    if (tid < 40) out[b * 40 + tid] = s[tid];
}

// ---------------------------------------------------------------------------
// Wasserstein means + logistic head -> y_pred [B,2]
// ---------------------------------------------------------------------------
__global__ void wasserstein_logistic(const float* __restrict__ fa3s, const float* __restrict__ r6s,
                                     const float* __restrict__ refs3s, const float* __restrict__ refs6s,
                                     const int* __restrict__ pred,
                                     const float* __restrict__ coef, const float* __restrict__ inter,
                                     float* __restrict__ ypred)
{
    __shared__ float srow[256];
    __shared__ float red[256];
    int b = blockIdx.x, t = threadIdx.x;
    srow[t] = fa3s[(size_t)b * 256 + t];
    __syncthreads();
    int p = pred[b];

    const float* R3 = refs3s + (size_t)p * 64 * 256;
    float acc = 0.0f;
    for (int s = 0; s < 64; ++s) acc += fabsf(srow[t] - R3[(size_t)s * 256 + t]);
    red[t] = acc; __syncthreads();
    for (int o = 128; o > 0; o >>= 1) { if (t < o) red[t] += red[t + o]; __syncthreads(); }
    float d3 = red[0] * (1.0f / (64.0f * 256.0f));
    __syncthreads();

    const float* R6 = refs6s + (size_t)p * 64 * 40;
    float acc6 = 0.0f;
    for (int i = t; i < 64 * 40; i += 256) { int k = i % 40; acc6 += fabsf(r6s[b * 40 + k] - R6[i]); }
    red[t] = acc6; __syncthreads();
    for (int o = 128; o > 0; o >>= 1) { if (t < o) red[t] += red[t + o]; __syncthreads(); }

    if (t == 0) {
        float d6 = red[0] * (1.0f / (64.0f * 40.0f));
        float z = d3 * coef[0] + d6 * coef[1] + inter[0];
        float pr = 1.0f / (1.0f + __expf(-z));
        ypred[b * 2 + 0] = 1.0f - pr;
        ypred[b * 2 + 1] = pr;
    }
}

// ---------------------------------------------------------------------------
// host orchestration
// ---------------------------------------------------------------------------
extern "C" void kernel_launch(void* const* d_in, const int* in_sizes, int n_in,
                              void* d_out, int out_size, void* d_ws, size_t ws_size,
                              hipStream_t stream)
{
    (void)in_sizes; (void)n_in; (void)out_size; (void)ws_size;

    const float* x     = (const float*)d_in[0];
    const float* w1mu  = (const float*)d_in[1];
    const float* w1ls  = (const float*)d_in[2];
    const float* b1mu  = (const float*)d_in[3];
    const float* b1ls  = (const float*)d_in[4];
    const float* w2mu  = (const float*)d_in[5];
    const float* w2ls  = (const float*)d_in[6];
    const float* b2mu  = (const float*)d_in[7];
    const float* b2ls  = (const float*)d_in[8];
    const float* f1wmu = (const float*)d_in[9];
    const float* f1wls = (const float*)d_in[10];
    const float* f1bmu = (const float*)d_in[11];
    const float* f1bls = (const float*)d_in[12];
    const float* f2wmu = (const float*)d_in[13];
    const float* f2wls = (const float*)d_in[14];
    const float* f2bmu = (const float*)d_in[15];
    const float* f2bls = (const float*)d_in[16];
    const float* pmean = (const float*)d_in[17];
    const float* pproj = (const float*)d_in[18];
    const float* refs3 = (const float*)d_in[19];
    const float* refs6 = (const float*)d_in[20];
    const float* coef  = (const float*)d_in[21];
    const float* inter = (const float*)d_in[22];

    char* base = (char*)d_ws;
    size_t off = 0;
    auto alloc = [&](size_t bytes) -> void* {
        off = (off + 255) & ~(size_t)255;
        void* p = base + off;
        off += bytes;
        return p;
    };

    float*    w1f    = (float*)alloc(500 * 4);
    float*    b1f    = (float*)alloc(20 * 4);
    float*    b2f    = (float*)alloc(50 * 4);
    float*    f1bf   = (float*)alloc(500 * 4);
    float*    f2wf   = (float*)alloc(5000 * 4);
    float*    f2bf   = (float*)alloc(10 * 4);
    _Float16* w2t    = (_Float16*)alloc((size_t)64 * 512 * 2);
    _Float16* f1wt   = (_Float16*)alloc((size_t)512 * 800 * 2);
    _Float16* h1h    = (_Float16*)alloc((size_t)BATCH * 20 * 144 * 2);
    _Float16* imc    = (_Float16*)alloc((size_t)CHUNK * 64 * 512 * 2);
    float*    c2raw  = (float*)alloc((size_t)CHUNK * 64 * 64 * 4);
    _Float16* pooled = (_Float16*)alloc((size_t)BATCH * 800 * 2);
    float*    fc1raw = (float*)alloc((size_t)BATCH * 512 * 4);
    _Float16* fbuf   = (_Float16*)alloc((size_t)BATCH * 512 * 2);
    _Float16* feat   = (_Float16*)alloc((size_t)BATCH * 3200 * 2);
    _Float16* projt  = (_Float16*)alloc((size_t)256 * 12800 * 2);
    float*    pm     = (float*)alloc(256 * 4);
    float*    agg    = (float*)alloc(BATCH * 10 * 4);
    float*    lbuf   = (float*)alloc(BATCH * 10 * 4);
    float*    lastb  = (float*)alloc(BATCH * 10 * 4);
    float*    fa3    = (float*)alloc((size_t)BATCH * 256 * 4);
    float*    fa3s   = (float*)alloc((size_t)BATCH * 256 * 4);
    float*    refs3s = (float*)alloc((size_t)10 * 64 * 256 * 4);
    float*    refs6s = (float*)alloc((size_t)10 * 64 * 40 * 4);
    float*    r6s    = (float*)alloc((size_t)BATCH * 40 * 4);
    int*      pred   = (int*)alloc(BATCH * 4);

    float* ypred = (float*)d_out;            // [1024,2]
    float* outlg = (float*)d_out + 2048;     // [1024,10]

    auto cdiv = [](int a, int b) { return (a + b - 1) / b; };

    auto sample = [&](uint32_t seed) {
        sample_weights<<<cdiv(SAMPLE_TOT, 256), 256, 0, stream>>>(seed,
            w1mu, w1ls, b1mu, b1ls, w2mu, w2ls, b2mu, b2ls,
            f1wmu, f1wls, f1bmu, f1bls, f2wmu, f2wls, f2bmu, f2bls,
            w1f, b1f, w2t, b2f, f1wt, f1bf, f2wf, f2bf);
    };
    auto gemm = [&](const _Float16* A, const _Float16* Bt, float* C,
                    int M, int N, int K, int lda, int ldb, int ldc, int acc) {
        int waves = (M / 16) * (N / 64);
        wmma_gemm_f16f32<<<cdiv(waves, 4), 128, 0, stream>>>(A, Bt, C, M, N, K, lda, ldb, ldc, acc);
    };
    auto conv_front = [&](uint32_t seed, bool feats) {
        sample(seed);
        conv1_relu_pool<<<cdiv(BATCH * 20 * 144, 256), 256, 0, stream>>>(x, w1f, b1f, h1h);
        for (int ch = 0; ch < NCHUNK; ++ch) {
            im2col_conv2<<<cdiv(CHUNK * 64 * 20, 256), 256, 0, stream>>>(h1h, imc, ch);
            gemm(imc, w2t, c2raw, CHUNK * 64, 64, 512, 512, 512, 64, 0);
            if (feats)
                conv2_feat_epilogue<<<cdiv(CHUNK * 64 * 50, 256), 256, 0, stream>>>(c2raw, b2f, feat, ch);
            else
                conv2_pool_epilogue<<<cdiv(CHUNK * 50 * 16, 256), 256, 0, stream>>>(c2raw, b2f, pooled, ch);
        }
    };
    auto full_pass = [&](uint32_t seed, float* lg, bool addagg) {
        conv_front(seed, false);
        gemm(pooled, f1wt, fc1raw, BATCH, 512, 800, 800, 800, 512, 0);
        fc1_epilogue<<<cdiv(BATCH * 512, 256), 256, 0, stream>>>(fc1raw, f1bf, fbuf);
        fc2_bayes<<<cdiv(BATCH * 10, 256), 256, 0, stream>>>(fbuf, f2wf, f2bf, lg, addagg ? agg : nullptr);
    };

    // 20-pass prediction aggregation (last pass's logits kept)
    zero_f32<<<cdiv(BATCH * 10, 256), 256, 0, stream>>>(agg, BATCH * 10);
    for (uint32_t i = 0; i < 19; ++i) full_pass(i, lbuf, true);
    full_pass(19u, lastb, true);
    argmax10<<<cdiv(BATCH, 256), 256, 0, stream>>>(agg, pred);

    // PCA path: fa3 = sum_t feat_t @ proj_t  -  mean @ proj
    transpose_proj_f16<<<cdiv(12800 * 256, 256), 256, 0, stream>>>(pproj, projt);
    pca_mean_proj<<<1, 256, 0, stream>>>(pmean, pproj, pm);
    fa3_init<<<cdiv(BATCH * 256, 256), 256, 0, stream>>>(fa3, pm);
    for (int t = 0; t < 4; ++t) {
        conv_front(100u + (uint32_t)t, true);
        // Bt slice: columns t*3200 .. t*3200+3199 of each projt row
        gemm(feat, projt + (size_t)t * 3200, fa3, BATCH, 256, 3200, 3200, 12800, 256, 1);
    }

    // sorts + Wasserstein + logistic head
    sort_rows_256<<<640, 256, 0, stream>>>(refs3, refs3s);
    sort_rows_40 <<<640, 64, 0, stream>>>(refs6, refs6s);
    sort_rows_256<<<BATCH, 256, 0, stream>>>(fa3, fa3s);
    build_sort_res6<<<BATCH, 64, 0, stream>>>(lastb, r6s);
    wasserstein_logistic<<<BATCH, 256, 0, stream>>>(fa3s, r6s, refs3s, refs6s, pred, coef, inter, ypred);

    // final stochastic pass -> output logits
    full_pass(999u, outlg, false);
}
